// LSTMNet_10419590660326
// MI455X (gfx1250) — matrix operand order repaired and checked by reading, not compiled, not run on hardware
//
#include <hip/hip_runtime.h>
#include <hip/hip_bf16.h>

// ---------------- problem dims ----------------
#define T_STEPS 512
#define BATCH   128
#define IN0     256
#define HID     512
#define BH      (BATCH * HID)
#define OUTF    60

// ---------------- recurrent kernel geometry ----------------
// 32 persistent blocks; block owns 16 hidden columns -> 4 gates * 16 = 64 gate
// columns = 4 N-tiles of 16. 8 waves/block, wave owns 16 batch rows (M-tile).
#define NBLK    32
#define HC_PER  16
#define WSTRIDE (1024 + 8)                 // padded LDS row (elements): 4-bank row skew
#define SMEM_BYTES (64 * WSTRIDE * 2 + 64 * 4)

typedef __attribute__((ext_vector_type(16))) __bf16 v16bf;
typedef __attribute__((ext_vector_type(8)))  float  v8f;

union Frag16 { v16bf v; uint4 q[2]; };

__device__ __forceinline__ unsigned short f2bf(float x) {
  union { float f; unsigned u; } v; v.f = x;
  unsigned r = v.u + 0x7FFFu + ((v.u >> 16) & 1u);   // round-to-nearest-even
  return (unsigned short)(r >> 16);
}
__device__ __forceinline__ float sigf(float x) { return 1.0f / (1.0f + __expf(-x)); }

#if __has_builtin(__builtin_amdgcn_tanhf)
__device__ __forceinline__ float fast_tanh(float x) { return __builtin_amdgcn_tanhf(x); }
#else
__device__ __forceinline__ float fast_tanh(float x) { return 2.0f * sigf(2.0f * x) - 1.0f; }
#endif

// Grid-wide split barrier: atomic arrive + generation spin.
__device__ __forceinline__ void grid_barrier(unsigned* bar, unsigned nblk) {
  __syncthreads();
  if (threadIdx.x == 0) {
    __threadfence();  // release: make h stores visible device-wide
    unsigned gen = __hip_atomic_load(&bar[1], __ATOMIC_RELAXED, __HIP_MEMORY_SCOPE_AGENT);
    unsigned arrived =
        __hip_atomic_fetch_add(&bar[0], 1u, __ATOMIC_ACQ_REL, __HIP_MEMORY_SCOPE_AGENT) + 1u;
    if (arrived == nblk) {
      __hip_atomic_store(&bar[0], 0u, __ATOMIC_RELAXED, __HIP_MEMORY_SCOPE_AGENT);
      __hip_atomic_fetch_add(&bar[1], 1u, __ATOMIC_RELEASE, __HIP_MEMORY_SCOPE_AGENT);
    } else {
      while (__hip_atomic_load(&bar[1], __ATOMIC_ACQUIRE, __HIP_MEMORY_SCOPE_AGENT) == gen)
        __builtin_amdgcn_s_sleep(2);
    }
  }
  __syncthreads();
}

// ---------------- prep: fp32 -> bf16 input, reset barrier ----------------
__global__ void prep_kernel(const float* __restrict__ xin,
                            unsigned short* __restrict__ xbf,
                            unsigned* __restrict__ bar, long n) {
  long i = (long)blockIdx.x * blockDim.x + threadIdx.x;
  if (i == 0) { bar[0] = 0u; bar[1] = 0u; }
  long stride = (long)gridDim.x * blockDim.x;
  for (; i < n; i += stride) xbf[i] = f2bf(xin[i]);
}

// ---------------- persistent LSTM layer (compile-time IN / last-layer flag) ----------------
template <int IN, bool IS_LAST>
__global__ __launch_bounds__(256, 1) void lstm_layer_kernel(
    const unsigned short* __restrict__ x,   // (T,B,IN) bf16
    unsigned short* __restrict__ y,         // (T,B,H) bf16 (next layer input); null if last
    float* __restrict__ pooled,             // (B,H) f32 time-sum; last layer only
    const float* __restrict__ W_ih,         // (4H,IN) f32
    const float* __restrict__ W_hh,         // (4H,H)  f32
    const float* __restrict__ b_ih, const float* __restrict__ b_hh,
    const float* __restrict__ h0,  const float* __restrict__ c0,   // (B,H) this layer
    unsigned short* __restrict__ h_buf,     // 2*B*H bf16 ping-pong
    unsigned* __restrict__ bar)
{
  constexpr int KT = IN + HID;              // fused [x_t | h_{t-1}] reduction length

  extern __shared__ unsigned char smem[];
  unsigned short* Wl = (unsigned short*)smem;            // [64][WSTRIDE] bf16, K-major
  float* bias_s      = (float*)(smem + 64 * WSTRIDE * 2);

  const int tid  = threadIdx.x;
  const int wave = tid >> 5;
  const int lane = tid & 31;
  const int half = lane >> 4;
  const int nc   = lane & 15;          // N column within tile / C-matrix column
  const int m0   = wave * 16;          // wave's batch-row tile
  const int hc0  = blockIdx.x * HC_PER;

  // ---- stage this block's weight slice into LDS (once per layer), bf16 ----
  for (int n = wave; n < 64; n += 8) {
    int gt = n >> 4, r = n & 15;
    int row = gt * HID + hc0 + r;
    const float* wi = W_ih + (size_t)row * IN;
    const float* wh = W_hh + (size_t)row * HID;
    for (int k = lane; k < KT; k += 32)
      Wl[n * WSTRIDE + k] = f2bf(k < IN ? wi[k] : wh[k - IN]);
  }
  if (tid < 64) {
    int gt = tid >> 4, r = tid & 15;
    int row = gt * HID + hc0 + r;
    bias_s[tid] = b_ih[row] + b_hh[row];
  }

  // ---- init h ping-pong slot 0 (this block covers its 16 columns, all B rows) ----
  for (int i = tid; i < BATCH * HC_PER; i += 256) {
    int b = i >> 4, cc = hc0 + (i & 15);
    h_buf[(size_t)b * HID + cc] = f2bf(h0[(size_t)b * HID + cc]);
  }

  // ---- per-lane recurrent state: c for (16 rows x 16 cols) tile, WMMA C layout ----
  const int col = hc0 + nc;
  float creg[8], pacc[8];
  #pragma unroll
  for (int r = 0; r < 8; ++r) {
    creg[r] = c0[(size_t)(m0 + half * 8 + r) * HID + col];
    pacc[r] = 0.0f;
  }
  __syncthreads();
  float bv[4];
  #pragma unroll
  for (int g = 0; g < 4; ++g) bv[g] = bias_s[g * 16 + nc];

  grid_barrier(bar, NBLK);   // h_buf[0] fully initialized across all blocks

  const int arow = m0 + nc;  // A-fragment row for this lane (lanes 0-15 / 16-31 same rows)

  // loop-invariant bases, hoisted so every inner access is base + immediate:
  //   A: x row (advance by B*IN per step) and the two h ping-pong rows
  //   B: two LDS bases (gates 0/1 and gates 2/3) -> all 8 ds_load offsets < 64KB
  const unsigned short* xrow   = x + (size_t)arow * IN + half * 8;
  const unsigned short* hrd[2] = {
      h_buf + (size_t)arow * HID + half * 8,
      h_buf + (size_t)BH + (size_t)arow * HID + half * 8 };
  const unsigned short* wb01 = Wl + (size_t)nc * WSTRIDE + half * 8;
  const unsigned short* wb23 = wb01 + 2 * 16 * WSTRIDE;
  const size_t obase = (size_t)(m0 + half * 8) * HID + col;   // store base (+ r*HID)

  for (int t = 0; t < T_STEPS; ++t) {
    const int prv = t & 1, cur = prv ^ 1;
    const unsigned short* hrow = hrd[prv];

    v8f acc[4];
    #pragma unroll
    for (int g = 0; g < 4; ++g)
      #pragma unroll
      for (int r = 0; r < 8; ++r) acc[g][r] = bv[g];

    // ---- fused input+recurrent GEMM over K = [x_t | h_{t-1}] ----
    // Fully unrolled (compile-time KT); A fragment software-pipelined one chunk ahead.
    Frag16 a;
    a.q[0] = *(const uint4*)(xrow);
    a.q[1] = *(const uint4*)(xrow + 16);
    #pragma unroll
    for (int kc = 0; kc < KT; kc += 32) {
      Frag16 an;
      if (kc + 32 < KT) {
        const unsigned short* nsrc =
            (kc + 32 < IN) ? (xrow + kc + 32) : (hrow + (kc + 32 - IN));
        an.q[0] = *(const uint4*)(nsrc);
        an.q[1] = *(const uint4*)(nsrc + 16);
      }
      // all 4 B fragments (immediate DS offsets), then 4 back-to-back WMMAs
      Frag16 b0, b1, b2, b3;
      b0.q[0] = *(const uint4*)(wb01 + kc);
      b0.q[1] = *(const uint4*)(wb01 + kc + 16);
      b1.q[0] = *(const uint4*)(wb01 + 16 * WSTRIDE + kc);
      b1.q[1] = *(const uint4*)(wb01 + 16 * WSTRIDE + kc + 16);
      b2.q[0] = *(const uint4*)(wb23 + kc);
      b2.q[1] = *(const uint4*)(wb23 + kc + 16);
      b3.q[0] = *(const uint4*)(wb23 + 16 * WSTRIDE + kc);
      b3.q[1] = *(const uint4*)(wb23 + 16 * WSTRIDE + kc + 16);
      acc[0] = __builtin_amdgcn_wmma_f32_16x16x32_bf16(false, a.v, false, b0.v, (short)0, acc[0], false, false);
      acc[1] = __builtin_amdgcn_wmma_f32_16x16x32_bf16(false, a.v, false, b1.v, (short)0, acc[1], false, false);
      acc[2] = __builtin_amdgcn_wmma_f32_16x16x32_bf16(false, a.v, false, b2.v, (short)0, acc[2], false, false);
      acc[3] = __builtin_amdgcn_wmma_f32_16x16x32_bf16(false, a.v, false, b3.v, (short)0, acc[3], false, false);
      if (kc + 32 < KT) a = an;
    }

    // prefetch next timestep's x row while we do the elementwise tail
    if (t + 1 < T_STEPS) __builtin_prefetch(xrow + (size_t)BATCH * IN, 0, 1);

    // ---- elementwise gates; i/f/g/o for one (b,h) share lane+element across tiles ----
    unsigned short* hwr = h_buf + (size_t)cur * BH;
    #pragma unroll
    for (int r = 0; r < 8; ++r) {
      float iv = sigf(acc[0][r]);
      float fv = sigf(acc[1][r]);
      float gv = fast_tanh(acc[2][r]);
      float ov = sigf(acc[3][r]);
      float c  = fv * creg[r] + iv * gv;
      creg[r]  = c;
      float h  = ov * fast_tanh(c);
      unsigned short h16 = f2bf(h);
      size_t oi = obase + (size_t)r * HID;
      hwr[oi] = h16;
      if (!IS_LAST) y[(size_t)t * BH + oi] = h16;
      else          pacc[r] += h;
    }

    xrow += (size_t)BATCH * IN;
    grid_barrier(bar, NBLK);  // publish h_t before any block reads it at t+1
  }

  if (IS_LAST) {
    #pragma unroll
    for (int r = 0; r < 8; ++r)
      pooled[obase + (size_t)r * HID] = pacc[r];
  }
}

// ---------------- final FC + tanh (tiny: 128x60x512) ----------------
__global__ void fc_kernel(const float* __restrict__ pooled,
                          const float* __restrict__ Wfc,
                          const float* __restrict__ bfc,
                          float* __restrict__ out) {
  int idx = blockIdx.x * blockDim.x + threadIdx.x;
  if (idx >= BATCH * OUTF) return;
  int b = idx / OUTF, o = idx % OUTF;
  const float* pr = pooled + (size_t)b * HID;
  const float* wr = Wfc + (size_t)o * HID;
  float s = bfc[o];
  #pragma unroll 8
  for (int k = 0; k < HID; ++k) s = fmaf(pr[k], wr[k], s);
  out[idx] = tanhf(s);
}

// ---------------- workspace layout (bytes) ----------------
static const size_t SZ_X0   = (size_t)T_STEPS * BATCH * IN0 * 2;   //  33.5 MB
static const size_t SZ_Y    = (size_t)T_STEPS * BATCH * HID * 2;   //  67.1 MB each
static const size_t OFF_X0  = 0;
static const size_t OFF_Y1  = OFF_X0 + SZ_X0;
static const size_t OFF_Y2  = OFF_Y1 + SZ_Y;
static const size_t OFF_HB  = OFF_Y2 + SZ_Y;                       // 2*BH bf16
static const size_t OFF_PL  = OFF_HB + 2 * (size_t)BH * 2;         // BH f32
static const size_t OFF_BAR = OFF_PL + (size_t)BH * 4;             // 2 u32
// total ~168.3 MB

extern "C" void kernel_launch(void* const* d_in, const int* in_sizes, int n_in,
                              void* d_out, int out_size, void* d_ws, size_t ws_size,
                              hipStream_t stream) {
  (void)in_sizes; (void)n_in; (void)out_size; (void)ws_size;
  const float* input = (const float*)d_in[0];
  const float* h0    = (const float*)d_in[1];   // (3,B,H)
  const float* c0    = (const float*)d_in[2];   // (3,B,H)
  const float* Wih[3] = {(const float*)d_in[3],  (const float*)d_in[7],  (const float*)d_in[11]};
  const float* Whh[3] = {(const float*)d_in[4],  (const float*)d_in[8],  (const float*)d_in[12]};
  const float* bih[3] = {(const float*)d_in[5],  (const float*)d_in[9],  (const float*)d_in[13]};
  const float* bhh[3] = {(const float*)d_in[6],  (const float*)d_in[10], (const float*)d_in[14]};
  const float* Wfc   = (const float*)d_in[15];
  const float* bfc   = (const float*)d_in[16];

  char* ws = (char*)d_ws;
  unsigned short* x0    = (unsigned short*)(ws + OFF_X0);
  unsigned short* y1    = (unsigned short*)(ws + OFF_Y1);
  unsigned short* y2    = (unsigned short*)(ws + OFF_Y2);
  unsigned short* h_buf = (unsigned short*)(ws + OFF_HB);
  float*          pool  = (float*)(ws + OFF_PL);
  unsigned*       bar   = (unsigned*)(ws + OFF_BAR);

  // 1) input fp32 -> bf16, reset grid barrier
  long nx = (long)T_STEPS * BATCH * IN0;
  prep_kernel<<<4096, 256, 0, stream>>>(input, x0, bar, nx);

  // 2) three persistent LSTM layers (each: grid-sync per timestep)
  lstm_layer_kernel<IN0, false><<<NBLK, 256, SMEM_BYTES, stream>>>(
      x0, y1, pool, Wih[0], Whh[0], bih[0], bhh[0],
      h0 + 0 * (size_t)BH, c0 + 0 * (size_t)BH, h_buf, bar);
  lstm_layer_kernel<HID, false><<<NBLK, 256, SMEM_BYTES, stream>>>(
      y1, y2, pool, Wih[1], Whh[1], bih[1], bhh[1],
      h0 + 1 * (size_t)BH, c0 + 1 * (size_t)BH, h_buf, bar);
  lstm_layer_kernel<HID, true><<<NBLK, 256, SMEM_BYTES, stream>>>(
      y2, nullptr, pool, Wih[2], Whh[2], bih[2], bhh[2],
      h0 + 2 * (size_t)BH, c0 + 2 * (size_t)BH, h_buf, bar);

  // 3) pooled @ W_fc^T + b, tanh
  fc_kernel<<<(BATCH * OUTF + 255) / 256, 256, 0, stream>>>(pool, Wfc, bfc, (float*)d_out);
}
